// W2ContrastiveLoss_90967407330088
// MI455X (gfx1250) — matrix operand order repaired and checked by reading, not compiled, not run on hardware
//
#include <hip/hip_runtime.h>
#include <hip/hip_bf16.h>
#include <math.h>

// ---------------------------------------------------------------------------
// W2 contrastive loss, MI455X (gfx1250, wave32).
//   w2[i,j] = asq[i] + bsq[j] - 2*cross[i,j]
//   cross   = [vm | sqrt(vv)] @ [tm | sqrt(tv)]^T   (single GEMM, K=1024)
//   logits  = -temp * w2 ; temp = min(exp(log_temp), 100)
//   loss    = 0.5*(mean lse_row + mean lse_col) - mean diag(logits)
//
// Pipeline:
//   1) pack: f32 -> bf16 hi/lo split matrices (once; removes 32x redundant cvt)
//   2) GEMM on V_WMMA_F32_16X16X32_BF16, bf16x3 split (~fp32 accuracy),
//      double-buffered LDS staged via GLOBAL_LOAD_ASYNC_TO_LDS_B128 (ASYNCcnt),
//      fused logits epilogue
//   3) row/col online logsumexp over L2-resident logits, finalize scalar.
// Workspace requirement: ~96.4 MB.
// ---------------------------------------------------------------------------

#define BSZ 4096
#define DD  512
#define KK  1024          // 2*DD
#define BM  128
#define BN  128
#define BK  32
#define LDA 40            // BK + 8 bf16 pad: 80B row stride, 16B aligned, 0-conflict
#define MSZ (BM * LDA)    // elements per staged matrix

#if defined(__has_builtin)
#if __has_builtin(__builtin_amdgcn_global_load_async_to_lds_b128)
#define HAS_ASYNC_LDS 1
#endif
#endif

typedef __attribute__((ext_vector_type(16))) __bf16 v16bf;
typedef __attribute__((ext_vector_type(8)))  __bf16 v8bf;
typedef __attribute__((ext_vector_type(8)))  float  v8f;
typedef __attribute__((ext_vector_type(4)))  int    v4i;

union Frag { v16bf v; v8bf h[2]; };
union Pack4 { __bf16 b[4]; uint2 u; };

__device__ __forceinline__ void copy_b128_to_lds(const __bf16* g, __bf16* l) {
#ifdef HAS_ASYNC_LDS
  __builtin_amdgcn_global_load_async_to_lds_b128(
      (__attribute__((address_space(1))) v4i*)g,
      (__attribute__((address_space(3))) v4i*)l, 0, 0);
#else
  *(uint4*)l = *(const uint4*)g;
#endif
}

__device__ __forceinline__ void wait_lds_copies() {
#ifdef HAS_ASYNC_LDS
#if __has_builtin(__builtin_amdgcn_s_wait_asynccnt)
  __builtin_amdgcn_s_wait_asynccnt(0);
#else
  asm volatile("s_wait_asynccnt 0" ::: "memory");
#endif
#endif
}

// ---------------------------------------------------------------------------
// Kernel 0: pack [mean | sqrt(var)] into bf16 hi/lo matrices [BSZ][KK]
// ---------------------------------------------------------------------------
__global__ __launch_bounds__(256)
void w2_pack(const float* __restrict__ vm, const float* __restrict__ vv,
             const float* __restrict__ tm, const float* __restrict__ tv,
             __bf16* __restrict__ PAh, __bf16* __restrict__ PAl,
             __bf16* __restrict__ PBh, __bf16* __restrict__ PBl) {
  const int row = blockIdx.x;
  const int k = threadIdx.x * 4;          // 0..1020
  const bool sig = (k >= DD);
  const int koff = sig ? (k - DD) : k;
  {
    const float* src = sig ? vv : vm;
    float4 f = *(const float4*)(src + (size_t)row * DD + koff);
    if (sig) { f.x = sqrtf(f.x); f.y = sqrtf(f.y); f.z = sqrtf(f.z); f.w = sqrtf(f.w); }
    Pack4 ph, pl;
    ph.b[0] = (__bf16)f.x; pl.b[0] = (__bf16)(f.x - (float)ph.b[0]);
    ph.b[1] = (__bf16)f.y; pl.b[1] = (__bf16)(f.y - (float)ph.b[1]);
    ph.b[2] = (__bf16)f.z; pl.b[2] = (__bf16)(f.z - (float)ph.b[2]);
    ph.b[3] = (__bf16)f.w; pl.b[3] = (__bf16)(f.w - (float)ph.b[3]);
    *(uint2*)(PAh + (size_t)row * KK + k) = ph.u;
    *(uint2*)(PAl + (size_t)row * KK + k) = pl.u;
  }
  {
    const float* src = sig ? tv : tm;
    float4 f = *(const float4*)(src + (size_t)row * DD + koff);
    if (sig) { f.x = sqrtf(f.x); f.y = sqrtf(f.y); f.z = sqrtf(f.z); f.w = sqrtf(f.w); }
    Pack4 ph, pl;
    ph.b[0] = (__bf16)f.x; pl.b[0] = (__bf16)(f.x - (float)ph.b[0]);
    ph.b[1] = (__bf16)f.y; pl.b[1] = (__bf16)(f.y - (float)ph.b[1]);
    ph.b[2] = (__bf16)f.z; pl.b[2] = (__bf16)(f.z - (float)ph.b[2]);
    ph.b[3] = (__bf16)f.w; pl.b[3] = (__bf16)(f.w - (float)ph.b[3]);
    *(uint2*)(PBh + (size_t)row * KK + k) = ph.u;
    *(uint2*)(PBl + (size_t)row * KK + k) = pl.u;
  }
}

// ---------------------------------------------------------------------------
// Kernel 1: asq[r] = sum_d vm^2 + vv ; bsq[r] = sum_d tm^2 + tv
// ---------------------------------------------------------------------------
__global__ __launch_bounds__(256)
void w2_prep(const float* __restrict__ vm, const float* __restrict__ vv,
             const float* __restrict__ tm, const float* __restrict__ tv,
             float* __restrict__ asq, float* __restrict__ bsq) {
  const int r = blockIdx.x;
  const int tid = threadIdx.x;
  float sa = 0.f, sb = 0.f;
  for (int d = tid; d < DD; d += 256) {
    float a = vm[r * DD + d];
    sa += a * a + vv[r * DD + d];
    float b = tm[r * DD + d];
    sb += b * b + tv[r * DD + d];
  }
  __shared__ float ra[256], rb[256];
  ra[tid] = sa; rb[tid] = sb;
  __syncthreads();
  for (int s = 128; s > 0; s >>= 1) {
    if (tid < s) { ra[tid] += ra[tid + s]; rb[tid] += rb[tid + s]; }
    __syncthreads();
  }
  if (tid == 0) { asq[r] = ra[0]; bsq[r] = rb[0]; }
}

// ---------------------------------------------------------------------------
// Kernel 2: GEMM (bf16x3) + fused logits epilogue, double-buffered async LDS.
// Block: 128x128 tile, 256 threads = 8 waves in 2(M) x 4(N); wave = 64x32.
// ---------------------------------------------------------------------------
__global__ __launch_bounds__(256)
void w2_gemm_logits(const __bf16* __restrict__ PAh, const __bf16* __restrict__ PAl,
                    const __bf16* __restrict__ PBh, const __bf16* __restrict__ PBl,
                    const float* __restrict__ asq, const float* __restrict__ bsq,
                    const float* __restrict__ log_temp, float* __restrict__ L) {
  // smem[buf][mat][BM*LDA]; mats: 0=Ah 1=Al 2=Bh 3=Bl.  2*4*128*40*2B = 80KB
  __shared__ __bf16 smem[2 * 4 * MSZ];

  const int tid   = threadIdx.x;
  const int mbase = blockIdx.y * BM;
  const int nbase = blockIdx.x * BN;

  const int wave = tid >> 5;
  const int lane = tid & 31;
  const int wm = wave >> 2;              // 0..1 : M position
  const int wn = wave & 3;               // 0..3 : N position
  const int lm = lane & 15;
  const int khalfA = (lane >> 4) * 8;    // A frag K-runs [khalfA..+7],[khalfA+16..+23]
  const int khalfB = (lane >> 4) * 8;    // B frag K-runs [khalfB..+7],[khalfB+16..+23]

  v8f acc[4][2] = {};

  // staging: 512 b128 chunks per matrix tile -> 2 per thread; row=c>>2, seg=c&3
  const int r0 = tid >> 2,         s0 = tid & 3;
  const int r1 = (tid + 256) >> 2, s1 = (tid + 256) & 3;

  auto stage = [&](int ks, int buf) {
    const int kg = ks * BK;
    __bf16* dst = &smem[buf * 4 * MSZ];
    {
      const size_t ga = (size_t)(mbase + r0) * KK + kg + s0 * 8;
      const size_t gb = (size_t)(nbase + r0) * KK + kg + s0 * 8;
      const int la = r0 * LDA + s0 * 8;
      copy_b128_to_lds(PAh + ga, dst + 0 * MSZ + la);
      copy_b128_to_lds(PAl + ga, dst + 1 * MSZ + la);
      copy_b128_to_lds(PBh + gb, dst + 2 * MSZ + la);
      copy_b128_to_lds(PBl + gb, dst + 3 * MSZ + la);
    }
    {
      const size_t ga = (size_t)(mbase + r1) * KK + kg + s1 * 8;
      const size_t gb = (size_t)(nbase + r1) * KK + kg + s1 * 8;
      const int la = r1 * LDA + s1 * 8;
      copy_b128_to_lds(PAh + ga, dst + 0 * MSZ + la);
      copy_b128_to_lds(PAl + ga, dst + 1 * MSZ + la);
      copy_b128_to_lds(PBh + gb, dst + 2 * MSZ + la);
      copy_b128_to_lds(PBl + gb, dst + 3 * MSZ + la);
    }
  };

  stage(0, 0);
  wait_lds_copies();
  __syncthreads();

  for (int ks = 0; ks < KK / BK; ++ks) {
    const int cur = ks & 1;
    if (ks + 1 < KK / BK) stage(ks + 1, cur ^ 1);   // overlap copy(ks+1) with math(ks)

    const __bf16* Ahp = &smem[cur * 4 * MSZ + 0 * MSZ];
    const __bf16* Alp = &smem[cur * 4 * MSZ + 1 * MSZ];
    const __bf16* Bhp = &smem[cur * 4 * MSZ + 2 * MSZ];
    const __bf16* Blp = &smem[cur * 4 * MSZ + 3 * MSZ];

    Frag ah[4], al[4], bh[2], bl[2];
#pragma unroll
    for (int mt = 0; mt < 4; ++mt) {
      const int m = wm * 64 + mt * 16 + lm;
      ah[mt].h[0] = *(const v8bf*)&Ahp[m * LDA + khalfA];
      ah[mt].h[1] = *(const v8bf*)&Ahp[m * LDA + khalfA + 16];
      al[mt].h[0] = *(const v8bf*)&Alp[m * LDA + khalfA];
      al[mt].h[1] = *(const v8bf*)&Alp[m * LDA + khalfA + 16];
    }
#pragma unroll
    for (int nt = 0; nt < 2; ++nt) {
      const int n = wn * 32 + nt * 16 + lm;
      bh[nt].h[0] = *(const v8bf*)&Bhp[n * LDA + khalfB];
      bh[nt].h[1] = *(const v8bf*)&Bhp[n * LDA + khalfB + 16];
      bl[nt].h[0] = *(const v8bf*)&Blp[n * LDA + khalfB];
      bl[nt].h[1] = *(const v8bf*)&Blp[n * LDA + khalfB + 16];
    }

    // bf16x3 split: x*y ~= hi*hi + hi*lo + lo*hi
#pragma unroll
    for (int mt = 0; mt < 4; ++mt) {
#pragma unroll
      for (int nt = 0; nt < 2; ++nt) {
        acc[mt][nt] = __builtin_amdgcn_wmma_f32_16x16x32_bf16(
            false, ah[mt].v, false, bh[nt].v, (short)0, acc[mt][nt], false, false);
        acc[mt][nt] = __builtin_amdgcn_wmma_f32_16x16x32_bf16(
            false, ah[mt].v, false, bl[nt].v, (short)0, acc[mt][nt], false, false);
        acc[mt][nt] = __builtin_amdgcn_wmma_f32_16x16x32_bf16(
            false, al[mt].v, false, bh[nt].v, (short)0, acc[mt][nt], false, false);
      }
    }

    wait_lds_copies();
    __syncthreads();
  }

  // epilogue: logits = temp * (2*cross - asq - bsq)
  const float temp = fminf(__expf(log_temp[0]), 100.0f);
  const int rhalf = (lane >> 4) * 8;   // C/D layout: lanes>=16 hold rows M+8
#pragma unroll
  for (int mt = 0; mt < 4; ++mt) {
    float aq[8];
#pragma unroll
    for (int r = 0; r < 8; ++r)
      aq[r] = asq[mbase + wm * 64 + mt * 16 + rhalf + r];
#pragma unroll
    for (int nt = 0; nt < 2; ++nt) {
      const int col = nbase + wn * 32 + nt * 16 + lm;
      const float bq = bsq[col];
#pragma unroll
      for (int r = 0; r < 8; ++r) {
        const int row = mbase + wm * 64 + mt * 16 + rhalf + r;
        L[(size_t)row * BSZ + col] = temp * (2.0f * acc[mt][nt][r] - aq[r] - bq);
      }
    }
  }
}

// ---------------------------------------------------------------------------
// Kernel 3: per-row online logsumexp
// ---------------------------------------------------------------------------
__global__ __launch_bounds__(256)
void w2_row_lse(const float* __restrict__ L, float* __restrict__ lse_row) {
  const int r = blockIdx.x;
  const int tid = threadIdx.x;
  float m = -INFINITY, s = 0.f;
  for (int j = tid; j < BSZ; j += 256) {
    const float x = L[(size_t)r * BSZ + j];
    if (x > m) { s = s * __expf(m - x) + 1.0f; m = x; }
    else       { s += __expf(x - m); }
  }
  __shared__ float sm[256], ss[256];
  sm[tid] = m; ss[tid] = s;
  __syncthreads();
  for (int k = 128; k > 0; k >>= 1) {
    if (tid < k) {
      const float m1 = sm[tid], s1 = ss[tid];
      const float m2 = sm[tid + k], s2 = ss[tid + k];
      const float M = fmaxf(m1, m2);
      sm[tid] = M;
      ss[tid] = s1 * __expf(m1 - M) + s2 * __expf(m2 - M);
    }
    __syncthreads();
  }
  if (tid == 0) lse_row[r] = sm[0] + logf(ss[0]);
}

// ---------------------------------------------------------------------------
// Kernel 4: per-column online logsumexp partials (8 row-chunks, coalesced)
// ---------------------------------------------------------------------------
__global__ __launch_bounds__(256)
void w2_col_partial(const float* __restrict__ L,
                    float* __restrict__ cpm, float* __restrict__ cps) {
  const int j = blockIdx.x * 256 + threadIdx.x;
  const int chunk = blockIdx.y;
  float m = -INFINITY, s = 0.f;
  const int i0 = chunk * (BSZ / 8);
  for (int i = i0; i < i0 + BSZ / 8; ++i) {
    const float x = L[(size_t)i * BSZ + j];
    if (x > m) { s = s * __expf(m - x) + 1.0f; m = x; }
    else       { s += __expf(x - m); }
  }
  cpm[chunk * BSZ + j] = m;
  cps[chunk * BSZ + j] = s;
}

// ---------------------------------------------------------------------------
// Kernel 5: combine column partials, diagonal, final scalar loss
// ---------------------------------------------------------------------------
__global__ __launch_bounds__(256)
void w2_finalize(const float* __restrict__ L, const float* __restrict__ lse_row,
                 const float* __restrict__ cpm, const float* __restrict__ cps,
                 float* __restrict__ out) {
  const int tid = threadIdx.x;
  float s_row = 0.f, s_col = 0.f, s_diag = 0.f;
  for (int i = tid; i < BSZ; i += 256) {
    s_row  += lse_row[i];
    s_diag += L[(size_t)i * BSZ + i];
    float m = -INFINITY, s = 0.f;
#pragma unroll
    for (int c = 0; c < 8; ++c) {
      const float m2 = cpm[c * BSZ + i], s2 = cps[c * BSZ + i];
      const float M = fmaxf(m, m2);
      s = s * __expf(m - M) + s2 * __expf(m2 - M);
      m = M;
    }
    s_col += m + logf(s);
  }
  __shared__ float r0[256], r1[256], r2[256];
  r0[tid] = s_row; r1[tid] = s_col; r2[tid] = s_diag;
  __syncthreads();
  for (int k = 128; k > 0; k >>= 1) {
    if (tid < k) { r0[tid] += r0[tid + k]; r1[tid] += r1[tid + k]; r2[tid] += r2[tid + k]; }
    __syncthreads();
  }
  if (tid == 0)
    out[0] = 0.5f * (r0[0] + r1[0]) * (1.0f / (float)BSZ) - r2[0] * (1.0f / (float)BSZ);
}

// ---------------------------------------------------------------------------
extern "C" void kernel_launch(void* const* d_in, const int* in_sizes, int n_in,
                              void* d_out, int out_size, void* d_ws, size_t ws_size,
                              hipStream_t stream) {
  (void)in_sizes; (void)n_in; (void)out_size; (void)ws_size;
  const float* vm = (const float*)d_in[0];
  const float* vv = (const float*)d_in[1];
  const float* tm = (const float*)d_in[2];
  const float* tv = (const float*)d_in[3];
  const float* lt = (const float*)d_in[4];
  float* out = (float*)d_out;

  char* base = (char*)d_ws;
  const size_t LBYTES = (size_t)BSZ * BSZ * sizeof(float);   // 64 MB
  const size_t PBYTES = (size_t)BSZ * KK * sizeof(__bf16);   // 8 MB each
  float*  L   = (float*)base;
  __bf16* PAh = (__bf16*)(base + LBYTES);
  __bf16* PAl = (__bf16*)(base + LBYTES + 1 * PBYTES);
  __bf16* PBh = (__bf16*)(base + LBYTES + 2 * PBYTES);
  __bf16* PBl = (__bf16*)(base + LBYTES + 3 * PBYTES);
  float* asq     = (float*)(base + LBYTES + 4 * PBYTES);
  float* bsq     = asq + BSZ;
  float* lse_row = bsq + BSZ;
  float* cpm     = lse_row + BSZ;      // 8*BSZ
  float* cps     = cpm + 8 * BSZ;      // 8*BSZ

  w2_pack<<<BSZ, 256, 0, stream>>>(vm, vv, tm, tv, PAh, PAl, PBh, PBl);
  w2_prep<<<BSZ, 256, 0, stream>>>(vm, vv, tm, tv, asq, bsq);

  dim3 gg(BSZ / BN, BSZ / BM);
  w2_gemm_logits<<<gg, 256, 0, stream>>>(PAh, PAl, PBh, PBl, asq, bsq, lt, L);

  w2_row_lse<<<BSZ, 256, 0, stream>>>(L, lse_row);

  dim3 gc(BSZ / 256, 8);
  w2_col_partial<<<gc, 256, 0, stream>>>(L, cpm, cps);

  w2_finalize<<<1, 256, 0, stream>>>(L, lse_row, cpm, cps, out);
}
